// MCSwinTransformerBlock_8813272891957
// MI455X (gfx1250) — matrix operand order repaired and checked by reading, not compiled, not run on hardware
//
#include <hip/hip_runtime.h>
#include <hip/hip_bf16.h>

typedef _Float16 half_t;
typedef _Float16 v16h __attribute__((ext_vector_type(16)));
typedef _Float16 v8h  __attribute__((ext_vector_type(8)));
typedef _Float16 v4h  __attribute__((ext_vector_type(4)));
typedef float    v8f  __attribute__((ext_vector_type(8)));

#define BATCH  32
#define NTOK   4096
#define CH     128
#define NHEAD  4
#define WSZ    64
#define SHIFT_ 32
#define NWIN   64
#define HID_   512

// D = A(16x32 f16) * B(32x16 f16) + C(16x16 f32)
__device__ __forceinline__ v8f wmma32(v16h a, v16h b, v8f c) {
  return __builtin_amdgcn_wmma_f32_16x16x32_f16(false, a, false, b, (short)0, c,
                                                false, false);
}

// A-fragment: row-major source, lane holds row M=lane%16,
// halves 0..7  = K (lane/16)*8 + 0..7,  halves 8..15 = K 16+(lane/16)*8 + 0..7
__device__ __forceinline__ v16h ldA(const half_t* base, int row, int ld,
                                    int kbase, int lane) {
  const half_t* p = base + row * ld + kbase + ((lane >> 4) << 3);
  v8h lo = *(const v8h*)p;
  v8h hi = *(const v8h*)(p + 16);
  v16h a;
#pragma unroll
  for (int i = 0; i < 8; ++i) { a[i] = lo[i]; a[i + 8] = hi[i]; }
  return a;
}

// B-fragment: lane holds column N (=ncol),
// halves 0..7 = K (lane/16)*8 + 0..7, halves 8..15 = K 16+(lane/16)*8 + 0..7
__device__ __forceinline__ v16h ldB(const half_t* base, int ncol, int ld,
                                    int kbase, int lane) {
  const half_t* p = base + (size_t)ncol * ld + kbase + ((lane >> 4) << 3);
  v8h lo = *(const v8h*)p;
  v8h hi = *(const v8h*)(p + 16);
  v16h b;
#pragma unroll
  for (int i = 0; i < 8; ++i) { b[i] = lo[i]; b[i + 8] = hi[i]; }
  return b;
}

// ---------------------------------------------------------------------------
// Weight conversion f32 -> f16 (once per launch)
// ---------------------------------------------------------------------------
__global__ void convert_w_kernel(const float* __restrict__ qkvw,
                                 const float* __restrict__ projw,
                                 const float* __restrict__ fc1w,
                                 const float* __restrict__ fc2w,
                                 half_t* qw, half_t* pw, half_t* f1, half_t* f2) {
  int i = blockIdx.x * blockDim.x + threadIdx.x;          // 0 .. 196607
  if (i < 49152) qw[i] = (half_t)qkvw[i];
  int j = i - 49152;
  if (j >= 0 && j < 16384) pw[j] = (half_t)projw[j];
  int k = i - 65536;
  if (k >= 0 && k < 65536) f1[k] = (half_t)fc1w[k];
  int l = i - 131072;
  if (l >= 0 && l < 65536) f2[l] = (half_t)fc2w[l];
}

// ---------------------------------------------------------------------------
// Fused: LN1 + roll(-32) + window QKV + attention + proj + residual -> x1
// one block = one window (64 tokens x 128 ch), 256 threads = 8 wave32s
// ---------------------------------------------------------------------------
__global__ __launch_bounds__(256) void swin_attn_kernel(
    const float* __restrict__ x, const float* __restrict__ amask,
    const float* __restrict__ ln1g, const float* __restrict__ ln1b,
    const half_t* __restrict__ qkvw, const float* __restrict__ qkvb,
    const float* __restrict__ pos, const half_t* __restrict__ projw,
    const float* __restrict__ projb, float* __restrict__ x1) {
  __shared__ half_t sXL[WSZ * CH];           // LN'd input; reused for attn-out
  __shared__ half_t sQ[WSZ * CH];            // scaled Q, row-major
  __shared__ half_t sK[WSZ * CH];            // K, row-major
  __shared__ half_t sVT[CH * WSZ];           // V transposed (chan, token)
  __shared__ half_t sP[NHEAD * WSZ * WSZ];   // softmax probs, per head

  const int blk = blockIdx.x;
  const int b   = blk >> 6;
  const int win = blk & 63;
  const int tid = threadIdx.x;
  const int w   = tid >> 5;
  const int lane = tid & 31;
  const int lh   = lane >> 4;
  const int ln16 = lane & 15;

  // ---- Stage 1: shifted load + LayerNorm -> sXL -------------------------
  {
    const float4 g4 = *(const float4*)(ln1g + lane * 4);
    const float4 bb4 = *(const float4*)(ln1b + lane * 4);
#pragma unroll
    for (int i = 0; i < 8; ++i) {
      int row = w * 8 + i;
      int t = (win * WSZ + row + SHIFT_) & (NTOK - 1);
      const float* xr = x + ((size_t)b * NTOK + t) * CH;
      float4 v = *(const float4*)(xr + lane * 4);
      float s = v.x + v.y + v.z + v.w;
      float ss = v.x * v.x + v.y * v.y + v.z * v.z + v.w * v.w;
#pragma unroll
      for (int m = 16; m >= 1; m >>= 1) {
        s += __shfl_xor(s, m, 32);
        ss += __shfl_xor(ss, m, 32);
      }
      float mu = s * (1.0f / CH);
      float rstd = rsqrtf(ss * (1.0f / CH) - mu * mu + 1e-5f);
      v4h o;
      o[0] = (half_t)((v.x - mu) * rstd * g4.x + bb4.x);
      o[1] = (half_t)((v.y - mu) * rstd * g4.y + bb4.y);
      o[2] = (half_t)((v.z - mu) * rstd * g4.z + bb4.z);
      o[3] = (half_t)((v.w - mu) * rstd * g4.w + bb4.w);
      *(v4h*)(sXL + row * CH + lane * 4) = o;
    }
  }
  __syncthreads();

  // ---- Stage 2: QKV GEMM (64x128)@(128x384), route to sQ/sK/sVT ---------
  // wave w owns N-tiles {w, w+8, w+16}: j selects Q/K/V statically.
  // B loaded once per (nt,kk) and reused over the 4 M-tiles (acc[4]).
  {
    const float qscale = 0.17677669529663689f;  // 32^-0.5
#pragma unroll
    for (int j = 0; j < 3; ++j) {
      int nt = w + j * 8;
      int nglob = nt * 16 + ln16;
      float bias = qkvb[nglob];
      v8f acc[4] = {};
#pragma unroll
      for (int kk = 0; kk < 4; ++kk) {
        v16h bm = ldB(qkvw, nglob, CH, kk * 32, lane);
#pragma unroll
        for (int mt = 0; mt < 4; ++mt) {
          v16h a = ldA(sXL, mt * 16 + ln16, CH, kk * 32, lane);
          acc[mt] = wmma32(a, bm, acc[mt]);
        }
      }
      if (j == 0) {  // Q (fold in softmax scale)
#pragma unroll
        for (int mt = 0; mt < 4; ++mt)
#pragma unroll
          for (int r = 0; r < 8; ++r)
            sQ[(mt * 16 + lh * 8 + r) * CH + nglob] =
                (half_t)((acc[mt][r] + bias) * qscale);
      } else if (j == 1) {  // K
#pragma unroll
        for (int mt = 0; mt < 4; ++mt)
#pragma unroll
          for (int r = 0; r < 8; ++r)
            sK[(mt * 16 + lh * 8 + r) * CH + (nglob - CH)] =
                (half_t)(acc[mt][r] + bias);
      } else {  // V stored transposed: contiguous 8-token b128 store per lane
#pragma unroll
        for (int mt = 0; mt < 4; ++mt) {
          v8h pk;
#pragma unroll
          for (int r = 0; r < 8; ++r) pk[r] = (half_t)(acc[mt][r] + bias);
          *(v8h*)(&sVT[(nglob - 2 * CH) * WSZ + mt * 16 + lh * 8]) = pk;
        }
      }
    }
  }
  __syncthreads();

  // ---- Stage 3: S = Q K^T + pos + mask, in-register softmax -> sP -------
  {
    const int h = w >> 1;             // head
    const int mr0 = (w & 1) * 32;     // 32 rows per wave
    v8f sacc[2][4];
#pragma unroll
    for (int nt = 0; nt < 4; ++nt) {
      v16h bm = ldB(sK, nt * 16 + ln16, CH, h * 32, lane);
#pragma unroll
      for (int mt2 = 0; mt2 < 2; ++mt2) {
        v8f acc = {};
        v16h a = ldA(sQ, mr0 + mt2 * 16 + ln16, CH, h * 32, lane);
        acc = wmma32(a, bm, acc);
#pragma unroll
        for (int r = 0; r < 8; ++r) {
          int M = mr0 + mt2 * 16 + lh * 8 + r;
          int Nc = nt * 16 + ln16;
          acc[r] += pos[(h * WSZ + M) * WSZ + Nc] +
                    amask[((size_t)win * WSZ + M) * WSZ + Nc];
        }
        sacc[mt2][nt] = acc;
      }
    }
    // softmax: each row lives in one 16-lane half at fixed vgpr index r
#pragma unroll
    for (int mt2 = 0; mt2 < 2; ++mt2) {
#pragma unroll
      for (int r = 0; r < 8; ++r) {
        float m = fmaxf(fmaxf(sacc[mt2][0][r], sacc[mt2][1][r]),
                        fmaxf(sacc[mt2][2][r], sacc[mt2][3][r]));
#pragma unroll
        for (int msk = 8; msk >= 1; msk >>= 1)
          m = fmaxf(m, __shfl_xor(m, msk, 32));
        float e0 = __expf(sacc[mt2][0][r] - m);
        float e1 = __expf(sacc[mt2][1][r] - m);
        float e2 = __expf(sacc[mt2][2][r] - m);
        float e3 = __expf(sacc[mt2][3][r] - m);
        float s = e0 + e1 + e2 + e3;
#pragma unroll
        for (int msk = 8; msk >= 1; msk >>= 1) s += __shfl_xor(s, msk, 32);
        float inv = 1.0f / s;
        int M = mr0 + mt2 * 16 + lh * 8 + r;
        half_t* pr = sP + (h * WSZ + M) * WSZ + ln16;
        pr[0] = (half_t)(e0 * inv);
        pr[16] = (half_t)(e1 * inv);
        pr[32] = (half_t)(e2 * inv);
        pr[48] = (half_t)(e3 * inv);
      }
    }
  }
  __syncthreads();

  // ---- Stage 4: O = P @ V  -> sXL (reused, f16 row-major 64x128) --------
  // wave w owns one 16-wide dim tile across all 4 M-tiles; B reused 4x.
  {
    const int nt = w;
    const int h = nt >> 1;  // dim = nt*16, head = dim/32
    v8f acc[4] = {};
#pragma unroll
    for (int ks = 0; ks < 2; ++ks) {
      v16h bm = ldB(sVT, nt * 16 + ln16, WSZ, ks * 32, lane);
#pragma unroll
      for (int mt = 0; mt < 4; ++mt) {
        v16h a = ldA(sP + h * WSZ * WSZ, mt * 16 + ln16, WSZ, ks * 32, lane);
        acc[mt] = wmma32(a, bm, acc[mt]);
      }
    }
#pragma unroll
    for (int mt = 0; mt < 4; ++mt)
#pragma unroll
      for (int r = 0; r < 8; ++r)
        sXL[(mt * 16 + lh * 8 + r) * CH + nt * 16 + ln16] =
            (half_t)acc[mt][r];
  }
  __syncthreads();

  // ---- Stage 5: proj + bias + shortcut (roll-back fused via index) ------
  {
    const int nt = w;
    const int nglob = nt * 16 + ln16;
    v8f acc[4] = {};
#pragma unroll
    for (int kk = 0; kk < 4; ++kk) {
      v16h bm = ldB(projw, nglob, CH, kk * 32, lane);
#pragma unroll
      for (int mt = 0; mt < 4; ++mt) {
        v16h a = ldA(sXL, mt * 16 + ln16, CH, kk * 32, lane);
        acc[mt] = wmma32(a, bm, acc[mt]);
      }
    }
    float pb = projb[nglob];
#pragma unroll
    for (int mt = 0; mt < 4; ++mt) {
#pragma unroll
      for (int r = 0; r < 8; ++r) {
        int row = mt * 16 + lh * 8 + r;
        int t = (win * WSZ + row + SHIFT_) & (NTOK - 1);
        size_t idx = ((size_t)b * NTOK + t) * CH + nglob;
        x1[idx] = acc[mt][r] + pb + x[idx];
      }
    }
  }
}

// ---------------------------------------------------------------------------
// Fused: LN2 + FC1 + GELU(erf) + FC2 + residual. One block = 64 tokens.
// ---------------------------------------------------------------------------
__global__ __launch_bounds__(256) void swin_mlp_kernel(
    const float* __restrict__ x1, const float* __restrict__ ln2g,
    const float* __restrict__ ln2b, const half_t* __restrict__ fc1w,
    const float* __restrict__ fc1b, const half_t* __restrict__ fc2w,
    const float* __restrict__ fc2b, float* __restrict__ out) {
  __shared__ half_t sX[64 * CH];     // LN2 output
  __shared__ half_t sH[64 * HID_];   // GELU(FC1) output
  const size_t row0 = (size_t)blockIdx.x * 64;
  const int tid = threadIdx.x;
  const int w = tid >> 5;
  const int lane = tid & 31;
  const int lh = lane >> 4;
  const int ln16 = lane & 15;

  // LN2
  {
    const float4 g4 = *(const float4*)(ln2g + lane * 4);
    const float4 bb4 = *(const float4*)(ln2b + lane * 4);
#pragma unroll
    for (int i = 0; i < 8; ++i) {
      int row = w * 8 + i;
      const float* xr = x1 + (row0 + row) * CH;
      float4 v = *(const float4*)(xr + lane * 4);
      float s = v.x + v.y + v.z + v.w;
      float ss = v.x * v.x + v.y * v.y + v.z * v.z + v.w * v.w;
#pragma unroll
      for (int m = 16; m >= 1; m >>= 1) {
        s += __shfl_xor(s, m, 32);
        ss += __shfl_xor(ss, m, 32);
      }
      float mu = s * (1.0f / CH);
      float rstd = rsqrtf(ss * (1.0f / CH) - mu * mu + 1e-5f);
      v4h o;
      o[0] = (half_t)((v.x - mu) * rstd * g4.x + bb4.x);
      o[1] = (half_t)((v.y - mu) * rstd * g4.y + bb4.y);
      o[2] = (half_t)((v.z - mu) * rstd * g4.z + bb4.z);
      o[3] = (half_t)((v.w - mu) * rstd * g4.w + bb4.w);
      *(v4h*)(sX + row * CH + lane * 4) = o;
    }
  }
  __syncthreads();

  // FC1 (64x128)@(128x512) + exact GELU; B reused over 4 M-tiles
#pragma unroll
  for (int j = 0; j < 4; ++j) {
    int nt = w * 4 + j;  // 32 N-tiles over 8 waves
    int nglob = nt * 16 + ln16;
    float bias = fc1b[nglob];
    v8f acc[4] = {};
#pragma unroll
    for (int kk = 0; kk < 4; ++kk) {
      v16h bm = ldB(fc1w, nglob, CH, kk * 32, lane);
#pragma unroll
      for (int mt = 0; mt < 4; ++mt) {
        v16h a = ldA(sX, mt * 16 + ln16, CH, kk * 32, lane);
        acc[mt] = wmma32(a, bm, acc[mt]);
      }
    }
#pragma unroll
    for (int mt = 0; mt < 4; ++mt) {
#pragma unroll
      for (int r = 0; r < 8; ++r) {
        int row = mt * 16 + lh * 8 + r;
        float vv = acc[mt][r] + bias;
        float gel = 0.5f * vv * (1.0f + erff(vv * 0.7071067811865475f));
        sH[row * HID_ + nglob] = (half_t)gel;
      }
    }
  }
  __syncthreads();

  // FC2 (64x512)@(512x128) + bias + residual; wave w owns N-tile w,
  // each B fragment loaded exactly once per block.
  {
    const int nglob = w * 16 + ln16;
    v8f acc[4] = {};
#pragma unroll
    for (int kk = 0; kk < 16; ++kk) {
      v16h bm = ldB(fc2w, nglob, HID_, kk * 32, lane);
#pragma unroll
      for (int mt = 0; mt < 4; ++mt) {
        v16h a = ldA(sH, mt * 16 + ln16, HID_, kk * 32, lane);
        acc[mt] = wmma32(a, bm, acc[mt]);
      }
    }
    float bias = fc2b[nglob];
#pragma unroll
    for (int mt = 0; mt < 4; ++mt) {
#pragma unroll
      for (int r = 0; r < 8; ++r) {
        int row = mt * 16 + lh * 8 + r;
        size_t idx = (row0 + row) * CH + nglob;
        out[idx] = acc[mt][r] + bias + x1[idx];
      }
    }
  }
}

// ---------------------------------------------------------------------------
extern "C" void kernel_launch(void* const* d_in, const int* in_sizes, int n_in,
                              void* d_out, int out_size, void* d_ws,
                              size_t ws_size, hipStream_t stream) {
  const float* x     = (const float*)d_in[0];
  const float* amask = (const float*)d_in[1];
  const float* ln1g  = (const float*)d_in[2];
  const float* ln1b  = (const float*)d_in[3];
  const float* qkvw  = (const float*)d_in[4];
  const float* qkvb  = (const float*)d_in[5];
  const float* pos   = (const float*)d_in[6];
  const float* projw = (const float*)d_in[7];
  const float* projb = (const float*)d_in[8];
  const float* ln2g  = (const float*)d_in[9];
  const float* ln2b  = (const float*)d_in[10];
  const float* fc1w  = (const float*)d_in[11];
  const float* fc1b  = (const float*)d_in[12];
  const float* fc2w  = (const float*)d_in[13];
  const float* fc2b  = (const float*)d_in[14];
  float* out = (float*)d_out;

  // workspace layout
  float* x1 = (float*)d_ws;  // B*N*C f32 = 64 MiB
  char* p = (char*)d_ws + (size_t)BATCH * NTOK * CH * sizeof(float);
  half_t* qw16 = (half_t*)p; p += (size_t)3 * CH * CH * sizeof(half_t);
  half_t* pw16 = (half_t*)p; p += (size_t)CH * CH * sizeof(half_t);
  half_t* f1w16 = (half_t*)p; p += (size_t)HID_ * CH * sizeof(half_t);
  half_t* f2w16 = (half_t*)p;

  convert_w_kernel<<<768, 256, 0, stream>>>(qkvw, projw, fc1w, fc2w, qw16, pw16,
                                            f1w16, f2w16);
  swin_attn_kernel<<<BATCH * NWIN, 256, 0, stream>>>(
      x, amask, ln1g, ln1b, qw16, qkvb, pos, pw16, projb, x1);
  swin_mlp_kernel<<<(BATCH * NTOK) / 64, 256, 0, stream>>>(
      x1, ln2g, ln2b, f1w16, fc1b, f2w16, fc2b, out);
}
